// RecurrentAgent_33586644254744
// MI455X (gfx1250) — matrix operand order repaired and checked by reading, not compiled, not run on hardware
//
#include <hip/hip_runtime.h>
#include <cstdint>
#include <cstddef>

// Problem constants (from reference): T=256, B=256, IN=512, H=256
#define T_STEPS 256
#define BATCH   256
#define IN_DIM  512
#define H_DIM   256
#define G4H     1024   // 4*H
#define SCAN_BLOCKS 256

typedef __attribute__((ext_vector_type(16))) __bf16 v16bf;
typedef __attribute__((ext_vector_type(8)))  float  v8f;

// float -> bf16, round-to-nearest-even
__device__ __forceinline__ __bf16 f2bf(float f) {
    unsigned u = __builtin_bit_cast(unsigned, f);
    u += 0x7FFFu + ((u >> 16) & 1u);
    unsigned short s = (unsigned short)(u >> 16);
    return __builtin_bit_cast(__bf16, s);
}

__device__ __forceinline__ float sigmoidf(float x) {
    return 1.0f / (1.0f + __expf(-x));
}

// CDNA5 async global->LDS copy (CDNA5 ISA): per-lane 16B,
// LDS dest offset in a VGPR, 64-bit global address in a VGPR pair.
// Tracked by ASYNCcnt.
__device__ __forceinline__ void async_copy_b128(unsigned lds_off, const void* gaddr) {
    asm volatile("global_load_async_to_lds_b128 %0, %1, off"
                 :: "v"(lds_off), "v"(gaddr) : "memory");
}
__device__ __forceinline__ void wait_async0() {
    asm volatile("s_wait_asynccnt 0" ::: "memory");
}

// Device-scope grid barrier for the persistent scan kernel (256 co-resident
// single-wave blocks). bar[0]=arrival count (monotonic), bar[1]=phase.
__device__ __forceinline__ void grid_barrier(int* cnt, int* phase, int t) {
    __builtin_amdgcn_fence(__ATOMIC_RELEASE, "agent");
    if (threadIdx.x == 0) {
        int old = __hip_atomic_fetch_add(cnt, 1, __ATOMIC_RELAXED, __HIP_MEMORY_SCOPE_AGENT);
        if (old == (t + 1) * SCAN_BLOCKS - 1) {
            __hip_atomic_store(phase, t + 1, __ATOMIC_RELEASE, __HIP_MEMORY_SCOPE_AGENT);
        } else {
            while (__hip_atomic_load(phase, __ATOMIC_RELAXED, __HIP_MEMORY_SCOPE_AGENT) <= t)
                __builtin_amdgcn_s_sleep(1);
        }
    }
    __builtin_amdgcn_fence(__ATOMIC_ACQUIRE, "agent");
}

// ---------------------------------------------------------------------------
// WMMA fragment layouts (CDNA5 ISA 7.12.2, wave32):
//  A (16x32 bf16, M x K): lane = 16*((k%16)/8) + m ; elem e = 8*(k/16) + (k%8)
//  B (32x16 bf16, K x N): lane = 16*((k%16)/8) + n ; elem e = 8*(k/16) + (k%8)
//  C/D (16x16 f32):       lane holds col n = lane%16, rows m = 8*(lane/16)+r
// Packed buffers store, per tile, 32 lanes x 16 bf16 contiguous, so GEMMs
// load one v16bf (32B) per lane per tile.
// ---------------------------------------------------------------------------

__global__ void pack_x_bf16(const float* __restrict__ X, __bf16* __restrict__ Apack) {
    size_t idx = (size_t)blockIdx.x * blockDim.x + threadIdx.x;   // TB*IN threads
    int m = (int)(idx / IN_DIM);
    int k = (int)(idx % IN_DIM);
    int tm = m >> 4, mm = m & 15;
    int tk = k >> 5, kk = k & 31;
    int lane = (((kk & 15) >> 3) << 4) + mm;
    int e    = ((kk >> 4) << 3) + (kk & 7);
    Apack[(((size_t)tm * (IN_DIM / 32) + tk) * 32 + lane) * 16 + e] = f2bf(X[idx]);
}

// W (4H x K f32 row-major) -> bf16 B-fragment order; B[k][n] = W[n][k].
__global__ void pack_w_bf16(const float* __restrict__ W, __bf16* __restrict__ Bpack, int K) {
    size_t idx = (size_t)blockIdx.x * blockDim.x + threadIdx.x;   // G4H*K threads
    int n = (int)(idx / K);
    int k = (int)(idx % K);
    int tn = n >> 4, nn = n & 15;
    int tk = k >> 5, kk = k & 31;
    int lane = (((kk & 15) >> 3) << 4) + nn;
    int e    = ((kk >> 4) << 3) + (kk & 7);
    Bpack[(((size_t)tn * (K / 32) + tk) * 32 + lane) * 16 + e] = f2bf(W[idx]);
}

__global__ void add_bias(const float* __restrict__ bi, const float* __restrict__ bh,
                         float* __restrict__ bias) {
    int i = blockIdx.x * blockDim.x + threadIdx.x;
    if (i < G4H) bias[i] = bi[i] + bh[i];
}

// h0 -> packed bf16 A-fragments; also reset the grid-barrier counters.
__global__ void init_state(const float* __restrict__ h0, __bf16* __restrict__ hpack,
                           int* __restrict__ bar) {
    int idx = blockIdx.x * blockDim.x + threadIdx.x;   // B*H threads
    if (idx < 2) bar[idx] = 0;
    int row = idx >> 8;          // H_DIM == 256
    int col = idx & 255;
    int tm = row >> 4;
    int tk = col >> 5, kk = col & 31;
    int lane = (((kk & 15) >> 3) << 4) + (row & 15);
    int e    = ((kk >> 4) << 3) + (kk & 7);
    hpack[(((size_t)tm * (H_DIM / 32) + tk) * 32 + lane) * 16 + e] = f2bf(h0[idx]);
}

// Phase 1: xproj = X @ W_ih^T + (b_ih + b_hh).  M=65536, N=1024, K=512.
// Block = 8 waves sharing one 64-col strip of W_ih staged in LDS via the
// async-to-LDS path; wave w computes M-tile mblk*8+w x 4 N-tiles.
__global__ void __launch_bounds__(256)
gemm_xproj(const __bf16* __restrict__ Apack, const __bf16* __restrict__ Bih,
           const float* __restrict__ bias, float* __restrict__ xproj) {
    __shared__ char Bsh[65536];   // 4 N-tiles x 16 K-tiles x 1KB
    int tid  = threadIdx.x;
    int wave = tid >> 5;
    int lane = tid & 31;
    int mblk = blockIdx.x >> 4;   // 0..511
    int ntb  = blockIdx.x & 15;   // 0..15
    int nlo  = lane & 15;

    // Cooperative async copy: 64KB contiguous strip (tiles tn=ntb*4..+3).
    {
        const char* src = (const char*)(Bih + (size_t)ntb * 32768);
        unsigned dst = (unsigned)(size_t)(&Bsh[0]);   // low 32 bits = LDS offset
#pragma unroll
        for (int i = 0; i < 16; ++i) {
            int off = (i * 256 + tid) * 16;
            async_copy_b128(dst + off, src + off);
        }
        wait_async0();
        __syncthreads();
    }

    int mt = mblk * 8 + wave;
    v8f acc[4];
#pragma unroll
    for (int j = 0; j < 4; ++j) {
        float bv = bias[(ntb * 4 + j) * 16 + nlo];
#pragma unroll
        for (int r = 0; r < 8; ++r) acc[j][r] = bv;
    }

    const v16bf* Af = ((const v16bf*)Apack) + ((size_t)mt * (IN_DIM / 32)) * 32 + lane;
    const v16bf* Bf = (const v16bf*)(&Bsh[0]);

#pragma unroll 4
    for (int kt = 0; kt < IN_DIM / 32; ++kt) {
        v16bf a = Af[(size_t)kt * 32];
#pragma unroll
        for (int j = 0; j < 4; ++j) {
            v16bf b = Bf[(j * 16 + kt) * 32 + lane];
            acc[j] = __builtin_amdgcn_wmma_f32_16x16x32_bf16(
                false, a, false, b, (short)0, acc[j], false, false);
        }
    }

    int rbase = mt * 16 + ((lane >> 4) << 3);
#pragma unroll
    for (int j = 0; j < 4; ++j) {
        int cb = (ntb * 4 + j) * 16 + nlo;
#pragma unroll
        for (int r = 0; r < 8; ++r)
            xproj[(size_t)(rbase + r) * G4H + cb] = acc[j][r];
    }
}

// Phase 2: persistent scan. 256 single-wave blocks (all co-resident); block
// (bm,hn) owns the 16x16 cell tile and computes all 4 gate tiles so the c/h
// update is lane-local. W_hh fragments staged once in LDS; c kept in VGPRs
// for the whole scan; h exchanged through a double-buffered packed global
// buffer with a device-scope barrier per step.
__global__ void __launch_bounds__(32)
lstm_scan(const __bf16* __restrict__ Bhh,
          const float*  __restrict__ xproj,
          const float*  __restrict__ done,
          const float*  __restrict__ c0,
          __bf16* __restrict__ hp0,
          __bf16* __restrict__ hp1,
          float* __restrict__ out_hidden,
          float* __restrict__ out_h,
          float* __restrict__ out_c,
          int* __restrict__ bar) {
    __shared__ char Bsh[32768];   // 4 gates x 8 K-tiles x 1KB
    int lane = threadIdx.x;
    int bm = blockIdx.x >> 4;
    int hn = blockIdx.x & 15;
    int nlo = lane & 15;

    // Stage the 4 gate strips of packed W_hh (tn = g*16+hn, 8KB each).
#pragma unroll
    for (int g = 0; g < 4; ++g) {
        const char* src = (const char*)(Bhh + ((size_t)(g * 16 + hn) * (H_DIM / 32)) * 32 * 16);
        unsigned dst = (unsigned)(size_t)(&Bsh[g * 8192]);
#pragma unroll
        for (int i = 0; i < 16; ++i) {
            int off = (i * 32 + lane) * 16;
            async_copy_b128(dst + off, src + off);
        }
    }
    wait_async0();

    int col = hn * 16 + nlo;
    int tkc = col >> 5;
    int kkc = col & 31;
    int laneh_base = (((kkc & 15) >> 3) << 4);
    int eh = ((kkc >> 4) << 3) + (kkc & 7);

    // c cells are block-private: keep in registers for the entire scan.
    float creg[8];
#pragma unroll
    for (int r = 0; r < 8; ++r) {
        int row = bm * 16 + ((lane >> 4) << 3) + r;
        creg[r] = c0[row * H_DIM + col];
    }

    for (int t = 0; t < T_STEPS; ++t) {
        const __bf16* hin = (t & 1) ? hp1 : hp0;
        __bf16* hout      = (t & 1) ? hp0 : hp1;

        // A-fragment lanes hold one batch row m = lane%16 -> mask = zeroing
        float dA = done[(size_t)t * BATCH + bm * 16 + nlo];
        v16bf zero = {};
        v16bf afrag[H_DIM / 32];
        const v16bf* Af = ((const v16bf*)hin) + ((size_t)bm * (H_DIM / 32)) * 32 + lane;
#pragma unroll
        for (int kt = 0; kt < H_DIM / 32; ++kt) {
            v16bf a = Af[kt * 32];
            afrag[kt] = (dA != 0.0f) ? zero : a;
        }

        v8f acc[4];
#pragma unroll
        for (int g = 0; g < 4; ++g) acc[g] = (v8f){};
#pragma unroll
        for (int g = 0; g < 4; ++g) {
            const v16bf* Bf = (const v16bf*)(&Bsh[g * 8192]);
#pragma unroll
            for (int kt = 0; kt < H_DIM / 32; ++kt) {
                acc[g] = __builtin_amdgcn_wmma_f32_16x16x32_bf16(
                    false, afrag[kt], false, Bf[kt * 32 + lane], (short)0, acc[g], false, false);
            }
        }

#pragma unroll
        for (int r = 0; r < 8; ++r) {
            int row = bm * 16 + ((lane >> 4) << 3) + r;
            float m = 1.0f - done[(size_t)t * BATCH + row];
            size_t xb = ((size_t)t * BATCH + row) * G4H + col;
            float gi = acc[0][r] + xproj[xb];
            float gf = acc[1][r] + xproj[xb + 256];
            float gg = acc[2][r] + xproj[xb + 512];
            float go = acc[3][r] + xproj[xb + 768];
            float cold = creg[r] * m;
            float cn = sigmoidf(gf) * cold + sigmoidf(gi) * tanhf(gg);
            float hv = sigmoidf(go) * tanhf(cn);
            creg[r] = cn;
            out_hidden[((size_t)t * BATCH + row) * H_DIM + col] = hv;
            hout[(((size_t)bm * (H_DIM / 32) + tkc) * 32 + (laneh_base + (row & 15))) * 16 + eh] =
                f2bf(hv);
            if (t == T_STEPS - 1) {
                out_h[row * H_DIM + col] = hv;
                out_c[row * H_DIM + col] = cn;
            }
        }

        if (t < T_STEPS - 1) grid_barrier(&bar[0], &bar[1], t);
    }
}

// ---------------------------------------------------------------------------
// Workspace layout (bytes, all 256-aligned):
//   [0,           268435456) xproj f32            (256 MB)
//   [268435456,   335544320) packed X bf16        ( 64 MB)
//   [335544320,   336592896) packed W_ih bf16     (  1 MB)
//   [336592896,   337117184) packed W_hh bf16     (0.5 MB)
//   [337117184,   337379328) packed h bf16 x2     (256 KB, double buffer)
//   [337379328,   337379584) grid barrier ints    (256 B)
//   [337379584,   337383680) bias f32             (  4 KB)
// ---------------------------------------------------------------------------
extern "C" void kernel_launch(void* const* d_in, const int* in_sizes, int n_in,
                              void* d_out, int out_size, void* d_ws, size_t ws_size,
                              hipStream_t stream) {
    const float* X    = (const float*)d_in[0];  // hidden (T*B, IN)
    const float* h0   = (const float*)d_in[1];
    const float* c0   = (const float*)d_in[2];
    const float* done = (const float*)d_in[3];
    const float* Wih  = (const float*)d_in[4];
    const float* Whh  = (const float*)d_in[5];
    const float* bih  = (const float*)d_in[6];
    const float* bhh  = (const float*)d_in[7];

    char* ws = (char*)d_ws;
    float*  xproj = (float*)(ws + 0ull);
    __bf16* Apack = (__bf16*)(ws + 268435456ull);
    __bf16* Bih   = (__bf16*)(ws + 335544320ull);
    __bf16* Bhh   = (__bf16*)(ws + 336592896ull);
    __bf16* hpA   = (__bf16*)(ws + 337117184ull);
    __bf16* hpB   = (__bf16*)(ws + 337117184ull + 131072ull);
    int*    bar   = (int*)  (ws + 337379328ull);
    float*  bias  = (float*)(ws + 337379584ull);

    float* out_hidden = (float*)d_out;
    float* out_h = out_hidden + (size_t)T_STEPS * BATCH * H_DIM;
    float* out_c = out_h + (size_t)BATCH * H_DIM;

    pack_x_bf16<<<(T_STEPS * BATCH * IN_DIM) / 256, 256, 0, stream>>>(X, Apack);
    pack_w_bf16<<<(G4H * IN_DIM) / 256, 256, 0, stream>>>(Wih, Bih, IN_DIM);
    pack_w_bf16<<<(G4H * H_DIM) / 256, 256, 0, stream>>>(Whh, Bhh, H_DIM);
    add_bias<<<G4H / 256, 256, 0, stream>>>(bih, bhh, bias);
    init_state<<<(BATCH * H_DIM) / 256, 256, 0, stream>>>(h0, hpA, bar);

    // 512 M-blocks x 16 N-strips, 8 waves per block
    gemm_xproj<<<8192, 256, 0, stream>>>(Apack, Bih, bias, xproj);

    // single persistent kernel replaces 256 per-step launches
    lstm_scan<<<SCAN_BLOCKS, 32, 0, stream>>>(Bhh, xproj, done, c0, hpA, hpB,
                                              out_hidden, out_h, out_c, bar);
}
